// WaveBlock_52158082843375
// MI455X (gfx1250) — compile-verified, matrix operand order checked by reading, not run on hardware
//
#include <hip/hip_runtime.h>
#include <hip/hip_bf16.h>

#define B_  8
#define C_  256
#define E_  512
#define T_  8192
#define GIN_ 256
#define R_  4

typedef __attribute__((ext_vector_type(16))) _Float16 v16h;
typedef __attribute__((ext_vector_type(8)))  _Float16 v8h;
typedef __attribute__((ext_vector_type(8)))  float    v8f;
typedef __attribute__((ext_vector_type(4)))  float    v4f;

// Load one 16x32 f16 A/B fragment half-row for this lane.
// lane<16  : K in {k0..k0+7} U {k0+16..k0+23} with k0 = kb*32
// lane>=16 : same + 8  (caller folds the +8*(lane>>4) into `base`)
__device__ inline v16h load_frag(const _Float16* __restrict__ base) {
    v8h lo = *(const v8h*)(base);
    v8h hi = *(const v8h*)(base + 16);
    return __builtin_shufflevector(lo, hi, 0, 1, 2, 3, 4, 5, 6, 7,
                                           8, 9, 10, 11, 12, 13, 14, 15);
}

// ---------------------------------------------------------------------------
// Kernel 1: causal dilated depthwise conv + channel LayerNorm -> f16 (B,T,C)
// block (32 t, 8 c-groups) ; grid (T/32, B)
// ---------------------------------------------------------------------------
__global__ __launch_bounds__(256) void conv_ln_kernel(
    const float* __restrict__ x,      // (B,C,T)
    const float* __restrict__ w,      // (C,3)
    const float* __restrict__ bias,   // (C)
    const float* __restrict__ gamma,  // (C)
    const float* __restrict__ beta,   // (C)
    _Float16* __restrict__ y16,       // (B,T,C)
    int dil)
{
    const int b  = blockIdx.y;
    const int tx = threadIdx.x;          // t within tile
    const int ty = threadIdx.y;          // channel group (32 channels each)
    const int t  = blockIdx.x * 32 + tx;

    __shared__ float yl[C_][33];
    __shared__ float red[2][8][32];

    const float* xb = x + (size_t)b * C_ * T_;
    float s = 0.f, s2 = 0.f;
#pragma unroll 4
    for (int cc = 0; cc < 32; ++cc) {
        const int c = ty * 32 + cc;
        const float* xc = xb + (size_t)c * T_;
        float acc = bias[c] + w[c * 3 + 0] * xc[t];
        if (t >= dil)     acc += w[c * 3 + 1] * xc[t - dil];
        if (t >= 2 * dil) acc += w[c * 3 + 2] * xc[t - 2 * dil];
        yl[c][tx] = acc;
        s += acc; s2 += acc * acc;
    }
    red[0][ty][tx] = s;
    red[1][ty][tx] = s2;
    __syncthreads();
    if (ty == 0) {
        float m = 0.f, v = 0.f;
#pragma unroll
        for (int j = 0; j < 8; ++j) { m += red[0][j][tx]; v += red[1][j][tx]; }
        m *= (1.f / (float)C_);
        v  = v * (1.f / (float)C_) - m * m;
        red[0][0][tx] = m;
        red[1][0][tx] = rsqrtf(v + 1e-5f);
    }
    __syncthreads();
    const float mu = red[0][0][tx];
    const float rs = red[1][0][tx];

    _Float16* out = y16 + ((size_t)b * T_ + t) * C_ + ty * 32;
#pragma unroll 4
    for (int cc = 0; cc < 32; ++cc) {
        const int c = ty * 32 + cc;
        out[cc] = (_Float16)((yl[c][tx] - mu) * rs * gamma[c] + beta[c]);
    }
}

// ---------------------------------------------------------------------------
// Kernel 2: fp32 -> f16 conversion of both pointwise weight matrices
// ---------------------------------------------------------------------------
__global__ __launch_bounds__(256) void cvt_w_kernel(
    const float* __restrict__ w1, const float* __restrict__ w2,
    _Float16* __restrict__ w1h, _Float16* __restrict__ w2h)
{
    const int i = blockIdx.x * 256 + threadIdx.x;
    if (i < E_ * C_) {
        w1h[i] = (_Float16)w1[i];
        w2h[i] = (_Float16)w2[i];
    }
}

// ---------------------------------------------------------------------------
// Kernel 3: LoRA projection vectors from g_out.
// aw per-batch layout: [0,1024) a_in1[c*R+r] | [1024,3072) a_out1[r*E+e]
//                    | [3072,5120) a_in2[e*R+r] | [5120,6144) a_out2[r*C+c]
// grid (48, B), block 128
// ---------------------------------------------------------------------------
__global__ __launch_bounds__(128) void lora_vec_kernel(
    const float* __restrict__ g,
    const float* __restrict__ p1ain, const float* __restrict__ p1aout,
    const float* __restrict__ p2ain, const float* __restrict__ p2aout,
    float* __restrict__ aw)
{
    const int b = blockIdx.y;
    const int tid = threadIdx.x;
    __shared__ float gs[GIN_];
    for (int i = tid; i < GIN_; i += 128) gs[i] = g[(size_t)b * GIN_ + i];
    __syncthreads();

    const int row = blockIdx.x * 128 + tid; // 0..6143
    const float* src;
    if (row < 1024)       src = p1ain  + (size_t)row * GIN_;
    else if (row < 3072)  src = p1aout + (size_t)(row - 1024) * GIN_;
    else if (row < 5120)  src = p2ain  + (size_t)(row - 3072) * GIN_;
    else                  src = p2aout + (size_t)(row - 5120) * GIN_;

    float s = 0.f;
#pragma unroll 8
    for (int i = 0; i < GIN_; ++i) s += src[i] * gs[i];
    aw[(size_t)b * 6144 + row] = s;
}

// ---------------------------------------------------------------------------
// Kernel 4: rank-4 projection  tq[b][r][t] = sum_c h16[b][t][c] * ain[c*4+r]
// grid (T/256, B), block 256
// ---------------------------------------------------------------------------
__global__ __launch_bounds__(256) void tvec_kernel(
    const _Float16* __restrict__ hsrc,   // (B,T,Cd)
    const float* __restrict__ aw, int aoff, int Cd,
    float* __restrict__ tq)              // (B,R,T)
{
    __shared__ float ain[E_ * R_];
    const int b = blockIdx.y;
    const int tid = threadIdx.x;
    for (int i = tid; i < Cd * R_; i += 256) ain[i] = aw[(size_t)b * 6144 + aoff + i];
    __syncthreads();

    const int t = blockIdx.x * 256 + tid;
    const _Float16* hp = hsrc + ((size_t)b * T_ + t) * Cd;
    float a0 = 0.f, a1 = 0.f, a2 = 0.f, a3 = 0.f;
    for (int c0 = 0; c0 < Cd; c0 += 8) {
        v8h hv = *(const v8h*)(hp + c0);
#pragma unroll
        for (int q = 0; q < 8; ++q) {
            const float h = (float)hv[q];
            const float* ap = &ain[(c0 + q) * R_];
            a0 += h * ap[0]; a1 += h * ap[1]; a2 += h * ap[2]; a3 += h * ap[3];
        }
    }
    float* o = tq + (size_t)b * R_ * T_ + t;
    o[0] = a0; o[T_] = a1; o[2 * T_] = a2; o[3 * T_] = a3;
}

// ---------------------------------------------------------------------------
// Kernel 5: GEMM1  h = gelu(W1 @ y + b1 + a_out1^T tq1)  (E x T per batch)
// block tile 128x128, 8 waves in 2(M)x4(N), wave tile 64x32 (4x2 WMMA tiles)
// ---------------------------------------------------------------------------
__global__ __launch_bounds__(256) void gemm1_kernel(
    const _Float16* __restrict__ w16,   // (E,C)
    const _Float16* __restrict__ y16,   // (B,T,C)
    const float* __restrict__ b1,       // (E)
    const float* __restrict__ aw,       // lora vectors
    const float* __restrict__ tq1,      // (B,R,T)
    _Float16* __restrict__ h16)         // (B,T,E)
{
    const int b    = blockIdx.z;
    const int tid  = threadIdx.x;
    const int wave = tid >> 5, lane = tid & 31;
    const int wm = wave >> 2, wn = wave & 3;
    const int m0 = blockIdx.y * 128 + wm * 64;
    const int n0 = blockIdx.x * 128 + wn * 32;
    const int r16 = lane & 15;
    const int hi8 = (lane >> 4) * 8;

    const _Float16* yb = y16 + (size_t)b * T_ * C_;

    v8f acc[4][2];
    {
        v8f z = {};
#pragma unroll
        for (int i = 0; i < 4; ++i)
#pragma unroll
            for (int j = 0; j < 2; ++j) acc[i][j] = z;
    }

    for (int kb = 0; kb < C_ / 32; ++kb) {
        const int k0 = kb * 32 + hi8;
        v16h af[4], bf[2];
#pragma unroll
        for (int i = 0; i < 4; ++i)
            af[i] = load_frag(w16 + (size_t)(m0 + i * 16 + r16) * C_ + k0);
#pragma unroll
        for (int j = 0; j < 2; ++j)
            bf[j] = load_frag(yb + (size_t)(n0 + j * 16 + r16) * C_ + k0);
#pragma unroll
        for (int i = 0; i < 4; ++i)
#pragma unroll
            for (int j = 0; j < 2; ++j)
                acc[i][j] = __builtin_amdgcn_wmma_f32_16x16x32_f16(
                    false, af[i], false, bf[j], (short)0, acc[i][j], false, false);
    }

    // Epilogue: bias + LoRA + exact GELU, LDS transpose, e-contiguous f16 store
    __shared__ __align__(16) _Float16 sh[8][16][16];
    const float* ao1 = aw + (size_t)b * 6144 + 1024;   // [r*E+e]
    const float* tqb = tq1 + (size_t)b * R_ * T_;

    for (int j = 0; j < 2; ++j) {
        const int t = n0 + j * 16 + r16;
        float tv[R_];
#pragma unroll
        for (int r = 0; r < R_; ++r) tv[r] = tqb[r * T_ + t];
        for (int i = 0; i < 4; ++i) {
#pragma unroll
            for (int v = 0; v < 8; ++v) {
                const int e = m0 + i * 16 + v + hi8;
                float val = acc[i][j][v] + b1[e];
#pragma unroll
                for (int r = 0; r < R_; ++r) val += ao1[r * E_ + e] * tv[r];
                val = 0.5f * val * (1.f + erff(val * 0.70710678118f));
                sh[wave][r16][v + hi8] = (_Float16)val;
            }
            __syncthreads();
            {
                v8h ov = *(const v8h*)&sh[wave][r16][hi8];
                _Float16* dst = h16 + ((size_t)b * T_ + (n0 + j * 16 + r16)) * E_
                                + m0 + i * 16 + hi8;
                *(v8h*)dst = ov;
            }
            __syncthreads();
        }
    }
}

// ---------------------------------------------------------------------------
// Kernel 6: GEMM2  x_out = x_res + W2 @ h + b2 + a_out2^T tq2  (C x T)
// same tiling; f32 LDS transpose -> t-contiguous vectorized (B,C,T) stores
// ---------------------------------------------------------------------------
__global__ __launch_bounds__(256) void gemm2_kernel(
    const _Float16* __restrict__ w16,   // (C,E)
    const _Float16* __restrict__ h16,   // (B,T,E)
    const float* __restrict__ b2,       // (C)
    const float* __restrict__ aw,
    const float* __restrict__ tq2,      // (B,R,T)
    const float* __restrict__ xres,     // (B,C,T)
    float* __restrict__ xout)           // (B,C,T)  (may alias xres)
{
    const int b    = blockIdx.z;
    const int tid  = threadIdx.x;
    const int wave = tid >> 5, lane = tid & 31;
    const int wm = wave >> 2, wn = wave & 3;
    const int m0 = blockIdx.y * 128 + wm * 64;
    const int n0 = blockIdx.x * 128 + wn * 32;
    const int r16 = lane & 15;
    const int hi8 = (lane >> 4) * 8;

    const _Float16* hb = h16 + (size_t)b * T_ * E_;

    v8f acc[4][2];
    {
        v8f z = {};
#pragma unroll
        for (int i = 0; i < 4; ++i)
#pragma unroll
            for (int j = 0; j < 2; ++j) acc[i][j] = z;
    }

    for (int kb = 0; kb < E_ / 32; ++kb) {
        const int k0 = kb * 32 + hi8;
        v16h af[4], bf[2];
#pragma unroll
        for (int i = 0; i < 4; ++i)
            af[i] = load_frag(w16 + (size_t)(m0 + i * 16 + r16) * E_ + k0);
#pragma unroll
        for (int j = 0; j < 2; ++j)
            bf[j] = load_frag(hb + (size_t)(n0 + j * 16 + r16) * E_ + k0);
#pragma unroll
        for (int i = 0; i < 4; ++i)
#pragma unroll
            for (int j = 0; j < 2; ++j)
                acc[i][j] = __builtin_amdgcn_wmma_f32_16x16x32_f16(
                    false, af[i], false, bf[j], (short)0, acc[i][j], false, false);
    }

    __shared__ __align__(16) float shf[8][16][16];
    const float* ao2 = aw + (size_t)b * 6144 + 5120;   // [r*C+c]
    const float* tqb = tq2 + (size_t)b * R_ * T_;

    for (int j = 0; j < 2; ++j) {
        const int t = n0 + j * 16 + r16;
        float tv[R_];
#pragma unroll
        for (int r = 0; r < R_; ++r) tv[r] = tqb[r * T_ + t];
        for (int i = 0; i < 4; ++i) {
#pragma unroll
            for (int v = 0; v < 8; ++v) {
                const int c = m0 + i * 16 + v + hi8;
                float val = acc[i][j][v] + b2[c];
#pragma unroll
                for (int r = 0; r < R_; ++r) val += ao2[r * C_ + c] * tv[r];
                shf[wave][r16][v + hi8] = val;
            }
            __syncthreads();
            {
                // lane handles row c = m0+i*16+r16, t-range [n0+j*16+hi8, +8)
                const size_t gidx = ((size_t)b * C_ + (m0 + i * 16 + r16)) * T_
                                    + n0 + j * 16 + hi8;
                v4f s0 = *(const v4f*)&shf[wave][r16][hi8];
                v4f s1 = *(const v4f*)&shf[wave][r16][hi8 + 4];
                v4f rr0 = *(const v4f*)(xres + gidx);
                v4f rr1 = *(const v4f*)(xres + gidx + 4);
                *(v4f*)(xout + gidx)     = s0 + rr0;
                *(v4f*)(xout + gidx + 4) = s1 + rr1;
            }
            __syncthreads();
        }
    }
}

// ---------------------------------------------------------------------------
extern "C" void kernel_launch(void* const* d_in, const int* in_sizes, int n_in,
                              void* d_out, int out_size, void* d_ws, size_t ws_size,
                              hipStream_t stream) {
    const float* x_in   = (const float*)d_in[0];
    const float* g      = (const float*)d_in[1];
    const float* dw     = (const float*)d_in[2];
    const float* db     = (const float*)d_in[3];
    const float* gam    = (const float*)d_in[4];
    const float* bet    = (const float*)d_in[5];
    const float* p1w    = (const float*)d_in[6];
    const float* p1b    = (const float*)d_in[7];
    const float* p1ain  = (const float*)d_in[8];
    const float* p1aout = (const float*)d_in[9];
    const float* p2w    = (const float*)d_in[10];
    const float* p2b    = (const float*)d_in[11];
    const float* p2ain  = (const float*)d_in[12];
    const float* p2aout = (const float*)d_in[13];
    float* xout = (float*)d_out;

    char* ws = (char*)d_ws;
    _Float16* y16 = (_Float16*)ws; ws += (size_t)B_ * T_ * C_ * 2;
    _Float16* h16 = (_Float16*)ws; ws += (size_t)B_ * T_ * E_ * 2;
    _Float16* w1h = (_Float16*)ws; ws += (size_t)E_ * C_ * 2;
    _Float16* w2h = (_Float16*)ws; ws += (size_t)E_ * C_ * 2;
    float* tq1 = (float*)ws; ws += (size_t)B_ * R_ * T_ * 4;
    float* tq2 = (float*)ws; ws += (size_t)B_ * R_ * T_ * 4;
    float* aw  = (float*)ws; ws += (size_t)B_ * 6144 * 4;

    const int dils[3] = {1, 3, 9};
    for (int L = 0; L < 3; ++L) {
        const float* xsrc = (L == 0) ? x_in : xout;

        conv_ln_kernel<<<dim3(T_ / 32, B_), dim3(32, 8), 0, stream>>>(
            xsrc, dw + (size_t)L * C_ * 3, db + L * C_, gam + L * C_, bet + L * C_,
            y16, dils[L]);

        cvt_w_kernel<<<dim3(E_ * C_ / 256), 256, 0, stream>>>(
            p1w + (size_t)L * E_ * C_, p2w + (size_t)L * C_ * E_, w1h, w2h);

        lora_vec_kernel<<<dim3(48, B_), 128, 0, stream>>>(
            g,
            p1ain  + (size_t)L * C_ * R_ * GIN_,
            p1aout + (size_t)L * E_ * R_ * GIN_,
            p2ain  + (size_t)L * E_ * R_ * GIN_,
            p2aout + (size_t)L * C_ * R_ * GIN_,
            aw);

        tvec_kernel<<<dim3(T_ / 256, B_), 256, 0, stream>>>(y16, aw, 0, C_, tq1);

        gemm1_kernel<<<dim3(T_ / 128, E_ / 128, B_), 256, 0, stream>>>(
            w1h, y16, p1b + L * E_, aw, tq1, h16);

        tvec_kernel<<<dim3(T_ / 256, B_), 256, 0, stream>>>(h16, aw, 3072, E_, tq2);

        gemm2_kernel<<<dim3(T_ / 128, C_ / 128, B_), 256, 0, stream>>>(
            w2h, h16, p2b + L * C_, aw, tq2, xsrc, xout);
    }
}
